// Conv2DShakeout_73873437491381
// MI455X (gfx1250) — matrix operand order repaired and checked by reading, not compiled, not run on hardware
//
#include <hip/hip_runtime.h>

// ---------------------------------------------------------------------------
// Shakeout Conv2d (3x3, stride 1, pad 1) for MI455X / gfx1250.
//   x:      (32, 256, 56, 56)  f32  NCHW
//   weight: (256, 256, 3, 3)   f32  OIHW
//   bias:   (256,)             f32
//   r:      (256, 256, 3, 3)   f32
//   out:    (32, 256, 56, 56)  f32
// Strategy: transform+pack weights to bf16 (WMMA-B layout), then implicit
// GEMM with v_wmma_f32_16x16x32_bf16. Each workgroup computes a 32-pixel
// (2 h-rows x 16 w) x 256-channel output tile so every B fragment load
// feeds two WMMAs. Tap loop pinned rolled (#pragma unroll 1): the 4-row
// variant made the scheduler spill to scratch inside the WMMA loop.
// ---------------------------------------------------------------------------

#define TAU  0.5f
#define CC_  0.7f
#define ITAU 0.5f

#define NB   32
#define CIN  256
#define KOUT 256
#define HH   56
#define WW   56
#define HW   (HH * WW)       // 3136
#define TAPS 9

typedef __attribute__((ext_vector_type(16))) __bf16 v16bf;
typedef __attribute__((ext_vector_type(8)))  float  v8f;

union Frag16 { v16bf v; uint4 q[2]; };        // 32 bytes: one WMMA A or B operand
union Acc8   { v8f v; float f[8]; float4 q[2]; };

__device__ __forceinline__ unsigned short f32_to_bf16(float f) {
    unsigned int u = __float_as_uint(f);
    u += 0x7fffu + ((u >> 16) & 1u);          // round to nearest even
    return (unsigned short)(u >> 16);
}

// ---------------------------------------------------------------------------
// Kernel 1: Shakeout transform + bf16 pack.
// Packed layout: wp[ ((tap*8 + c/32)*256 + k)*32 + (c%32) ]  (bf16)
// => a lane's B fragment (c 8h+{0..7,16..23} for one k, h = lane/16) is two 16-byte
//    chunks -> two global_load_b128, L2-resident across all workgroups.
// ---------------------------------------------------------------------------
__global__ __launch_bounds__(256) void shakeout_pack_kernel(
        const float* __restrict__ weight,
        const float* __restrict__ rmat,
        unsigned short* __restrict__ wp) {
    int i = blockIdx.x * 256 + threadIdx.x;           // over 256*256*9
    if (i >= KOUT * CIN * TAPS) return;
    int tap = i % TAPS;
    int c   = (i / TAPS) % CIN;
    int k   = i / (TAPS * CIN);

    float w  = weight[i];
    float rv = rmat[i];
    float sg = (w > 0.f) ? 1.f : ((w < 0.f) ? -1.f : 0.f);
    float wt;
    if (rv > TAU)      wt = CC_ * sg + ITAU * w;                 // imask=1
    else if (rv < TAU) wt = ITAU * (w + CC_ * TAU * sg);         // mask=1
    else               wt = ITAU * w;                            // measure zero

    int cc = c >> 5, ci = c & 31;
    wp[((size_t)(tap * 8 + cc) * KOUT + k) * 32 + ci] = f32_to_bf16(wt);
}

// ---------------------------------------------------------------------------
// Kernel 2: implicit-GEMM conv.
// Grid: 32(n) * 28(h-pairs) * 4(w-tiles of 16) = 3584 blocks, 256 threads
// (8 waves). LDS x tile: 4 h-rows x 18 w x 256 c, bf16, [row][w][c].
// Wave wid covers output channels [32*wid, 32*wid+32) for BOTH output rows:
// 4 accumulators, B fragments reused 2x, A fragments reused 2x.
// ---------------------------------------------------------------------------
__global__ __launch_bounds__(256, 2) void shakeout_conv_kernel(
        const float* __restrict__ x,
        const float* __restrict__ bias,
        const unsigned short* __restrict__ wp,
        float* __restrict__ out) {
    __shared__ unsigned short xs[4 * 18 * CIN];       // 18432 elems = 36864 B

    const int b  = blockIdx.x;
    const int wt = b & 3;                             // w-tile: w0 = 16*wt
    const int hp = (b >> 2) % (HH / 2);               // h-pair: h0 = 2*hp
    const int n  = b / (4 * (HH / 2));
    const int w0 = wt << 4;
    const int h0 = hp << 1;

    const int tid   = threadIdx.x;
    const int wid   = tid >> 5;                       // wave 0..7
    const int lane  = tid & 31;
    const int l16   = lane & 15;
    const int lhalf = lane >> 4;                      // 0: lanes 0-15, 1: 16-31

    // -------- stage x halo tile (f32 -> bf16) into LDS ---------------------
    // 18432 = 72 * 256: every thread runs exactly 72 iterations (no EXEC
    // divergence so later WMMA sees EXEC all-ones).
    const size_t xn = (size_t)n * CIN * HW;
    for (int idx = tid; idx < 4 * 18 * CIN; idx += 256) {
        int c   = idx / 72;
        int rem = idx - c * 72;
        int row = rem / 18;                           // 0..3  -> gh = h0-1+row
        int wi  = rem - row * 18;                     // 0..17 -> gw = w0-1+wi
        int gh  = h0 - 1 + row;
        int gw  = w0 - 1 + wi;
        float v = 0.f;
        if ((unsigned)gh < HH && (unsigned)gw < WW)
            v = x[xn + (size_t)c * HW + gh * WW + gw];
        xs[(row * 18 + wi) * CIN + c] = f32_to_bf16(v);
    }
    __syncthreads();

    // -------- main WMMA loops ---------------------------------------------
    v8f acc00 = {};   // out row h0,   k in [32*wid, 32*wid+16)
    v8f acc01 = {};   // out row h0,   k in [32*wid+16, 32*wid+32)
    v8f acc10 = {};   // out row h0+1, k lower half
    v8f acc11 = {};   // out row h0+1, k upper half

    // B fragment base for this lane: k = 32*wid + l16 (+16 for *1 accs),
    // c-halfword offset lhalf*8 within the 32-c chunk.
    const unsigned short* wb_lane = wp + ((size_t)(wid * 32 + l16) * 32 + lhalf * 8);

    #pragma unroll 1                                  // keep rolled: avoids spills
    for (int tap = 0; tap < TAPS; ++tap) {
        const int row = tap / 3;
        const int s   = tap - row * 3;
        // A fragments: pixel M = l16 -> LDS w index l16+s; output row m uses
        // LDS h-row (row+m). K chunk select per 16-bit A layout: lanes 0-15
        // take c {0..7,16..23}, lanes 16-31 take c {8..15,24..31}.
        const unsigned short* xa0 = &xs[((row + 0) * 18 + l16 + s) * CIN + lhalf * 8];
        const unsigned short* xa1 = &xs[((row + 1) * 18 + l16 + s) * CIN + lhalf * 8];
        const unsigned short* wb  = wb_lane + (size_t)tap * 8 * KOUT * 32;

        #pragma unroll
        for (int cc = 0; cc < 8; ++cc) {
            Frag16 a0, a1, b0, b1;
            const uint4* ap0 = (const uint4*)(xa0 + cc * 32);
            a0.q[0] = ap0[0];                         // c: base+0..7
            a0.q[1] = ap0[2];                         // c: base+16..23
            const uint4* ap1 = (const uint4*)(xa1 + cc * 32);
            a1.q[0] = ap1[0];
            a1.q[1] = ap1[2];

            const uint4* bp0 = (const uint4*)(wb + (size_t)cc * KOUT * 32);
            b0.q[0] = bp0[0];
            b0.q[1] = bp0[2];                         // c: base+0..7, base+16..23 (L2-resident)
            const uint4* bp1 = (const uint4*)(wb + (size_t)cc * KOUT * 32 + 16 * 32);
            b1.q[0] = bp1[0];
            b1.q[1] = bp1[2];

            acc00 = __builtin_amdgcn_wmma_f32_16x16x32_bf16(
                        false, a0.v, false, b0.v, (short)0, acc00, false, false);
            acc01 = __builtin_amdgcn_wmma_f32_16x16x32_bf16(
                        false, a0.v, false, b1.v, (short)0, acc01, false, false);
            acc10 = __builtin_amdgcn_wmma_f32_16x16x32_bf16(
                        false, a1.v, false, b0.v, (short)0, acc10, false, false);
            acc11 = __builtin_amdgcn_wmma_f32_16x16x32_bf16(
                        false, a1.v, false, b1.v, (short)0, acc11, false, false);
        }
    }

    // -------- epilogue: bias + store --------------------------------------
    // C/D layout: VGPR j, lanes 0-15 -> M=j ; lanes 16-31 -> M=j+8 ; N=l16.
    // M maps to w offset -> each lane's 8 floats are 8 consecutive w's.
    const int k0 = wid * 32 + l16;
    const int k1 = k0 + 16;
    const float bv0 = bias[k0];
    const float bv1 = bias[k1];

    Acc8 r00, r01, r10, r11;
    r00.v = acc00 + bv0;
    r01.v = acc01 + bv1;
    r10.v = acc10 + bv0;
    r11.v = acc11 + bv1;

    const int baseW = w0 + lhalf * 8;
    const size_t o00 = ((size_t)(n * KOUT + k0) * HH + h0) * WW + baseW;
    const size_t o10 = o00 + WW;                      // row h0+1
    const size_t o01 = o00 + (size_t)16 * HW;         // k upper half
    const size_t o11 = o01 + WW;

    if (w0 != 48) {                                   // full 16-wide tile
        *(float4*)(out + o00)     = r00.q[0];
        *(float4*)(out + o00 + 4) = r00.q[1];
        *(float4*)(out + o01)     = r01.q[0];
        *(float4*)(out + o01 + 4) = r01.q[1];
        *(float4*)(out + o10)     = r10.q[0];
        *(float4*)(out + o10 + 4) = r10.q[1];
        *(float4*)(out + o11)     = r11.q[0];
        *(float4*)(out + o11 + 4) = r11.q[1];
    } else {                                          // edge tile: w in [48,56)
        #pragma unroll
        for (int j = 0; j < 8; ++j) {
            if (baseW + j < WW) {
                out[o00 + j] = r00.f[j];
                out[o01 + j] = r01.f[j];
                out[o10 + j] = r10.f[j];
                out[o11 + j] = r11.f[j];
            }
        }
    }
}

// ---------------------------------------------------------------------------
extern "C" void kernel_launch(void* const* d_in, const int* in_sizes, int n_in,
                              void* d_out, int out_size, void* d_ws, size_t ws_size,
                              hipStream_t stream) {
    const float* x      = (const float*)d_in[0];
    const float* weight = (const float*)d_in[1];
    const float* bias   = (const float*)d_in[2];
    const float* rmat   = (const float*)d_in[3];
    float* out          = (float*)d_out;
    unsigned short* wp  = (unsigned short*)d_ws;      // 2304*256*2 B = 1.18 MB

    (void)in_sizes; (void)n_in; (void)out_size; (void)ws_size;

    // 1) Shakeout transform + bf16 pack (256*256*9 = 589824 elements)
    shakeout_pack_kernel<<<(KOUT * CIN * TAPS + 255) / 256, 256, 0, stream>>>(
        weight, rmat, wp);

    // 2) Implicit GEMM conv: 32 * 28 * 4 = 3584 workgroups of 256 threads
    shakeout_conv_kernel<<<NB * (HH / 2) * 4, 256, 0, stream>>>(x, bias, wp, out);
}
